// TransformerBlock_28449863369586
// MI455X (gfx1250) — matrix operand order repaired and checked by reading, not compiled, not run on hardware
//
#include <hip/hip_runtime.h>
#include <hip/hip_bf16.h>

// ---------------------------------------------------------------------------
// CDNA5 / gfx1250, wave32. Matmuls: v_wmma_f32_16x16x32_bf16, f32 accum.
// Data paths: 128-bit loads, async global->LDS (ASYNCcnt), TDM (TENSORcnt).
// ---------------------------------------------------------------------------

typedef __bf16 bf16;
typedef bf16  v16bf __attribute__((ext_vector_type(16)));
typedef bf16  v8bf  __attribute__((ext_vector_type(8)));
typedef bf16  v4bf  __attribute__((ext_vector_type(4)));
typedef float v8f   __attribute__((ext_vector_type(8)));
typedef unsigned int u32x4 __attribute__((ext_vector_type(4)));
typedef int   i32x4 __attribute__((ext_vector_type(4)));
typedef int   i32x8 __attribute__((ext_vector_type(8)));

#define Bsz   256
#define Ssz   300
#define Dm    64
#define DFFsz 256
#define NHn   2
#define DKn   32
#define SP    320   // S padded to a multiple of 32 (P@V K-dim)

static __device__ __forceinline__ v8f wmma_bf16(v16bf a, v16bf b, v8f c) {
  return __builtin_amdgcn_wmma_f32_16x16x32_bf16(false, a, false, b, (short)0, c,
                                                 false, false);
}

static __device__ __forceinline__ v16bf join16(v8bf lo, v8bf hi) {
  return __builtin_shufflevector(lo, hi, 0, 1, 2, 3, 4, 5, 6, 7,
                                 8, 9, 10, 11, 12, 13, 14, 15);
}

// A fragment 16x32 bf16 from a row-major tile (ld = K stride).
static __device__ __forceinline__ v16bf load_a_frag(const bf16* tile, int ld,
                                                    int lane, int kbase) {
  const bf16* p = tile + (lane & 15) * ld + kbase + ((lane >> 4) << 3);
  return join16(*(const v8bf*)p, *(const v8bf*)(p + 16));
}

// B fragment 32x16 from a TRANSPOSED [N][K] row-major matrix (ld = K stride).
static __device__ __forceinline__ v16bf load_bt_frag(const bf16* matT, int ld,
                                                     int lane, int kbase, int nbase) {
  const bf16* p = matT + (size_t)(nbase + (lane & 15)) * ld + kbase + ((lane >> 4) << 3);
  return join16(*(const v8bf*)p, *(const v8bf*)(p + 16));
}

// ---- CDNA5 async global->LDS (ASYNCcnt) ----------------------------------
static __device__ __forceinline__ uint32_t lds_off(const void* p) {
  return (uint32_t)(uintptr_t)p;  // LDS offset = low 32 bits of the flat address (ISA 10.2)
}
static __device__ __forceinline__ void async_b128(uint32_t lds, const void* g) {
  asm volatile("global_load_async_to_lds_b128 %0, %1, off"
               :: "v"(lds), "v"((uint64_t)(uintptr_t)g) : "memory");
}
static __device__ __forceinline__ void wait_async0() {
  asm volatile("s_wait_asynccnt 0x0" ::: "memory");
}

// ---- CDNA5 Tensor Data Mover: 2D tile, bf16, tile == tensor ---------------
// D# per CDNA5 ISA: group0 = {count|flags, lds_addr,
// global_addr[31:0], global_addr[56:32]|type=2}; group1 packs data_size=2B,
// tensor_dim0/1, tile_dim0/1, tensor_dim0_stride. Groups 2/3 unused (2D).
// This toolchain (clang-23/therock-10.0) uses the 6-arg builtin form.
static __device__ __forceinline__ void tdm_load_2d_bf16(uint32_t lds, const void* gptr,
                                                        uint32_t dim0, uint32_t dim1,
                                                        uint32_t stride0) {
  const uint64_t ga = (uint64_t)(uintptr_t)gptr;
  u32x4 g0 = { 1u,                                   // count=1 (valid user D#)
               lds,                                  // lds_addr  (bits 63:32)
               (uint32_t)ga,                         // global_addr[31:0]
               (uint32_t)((ga >> 32) & 0x01FFFFFFu) | (2u << 30) };  // [56:32]|type=2
  i32x8 g1 = { (int)(1u << 16),                      // data_size = 2 bytes
               (int)(dim0 << 16),                    // tensor_dim0[15:0] @ bits 63:48
               (int)(dim1 << 16),                    // tensor_dim1[15:0] @ bits 95:80
               (int)(dim0 << 16),                    // tile_dim0 @ bits 127:112
               (int)(dim1 & 0xFFFFu),                // tile_dim1 @ bits 143:128
               (int)stride0,                         // tensor_dim0_stride[31:0]
               0, 0 };
  i32x4 z4 = { 0, 0, 0, 0 };
  i32x8 z8 = { 0, 0, 0, 0, 0, 0, 0, 0 };
  __builtin_amdgcn_tensor_load_to_lds(g0, g1, z4, z4, z8, 0);
}

// ---------------------------------------------------------------------------
// Pass 0: weight f32 [K][N]  ->  bf16 [N][K] (transposed for B-fragments)
// ---------------------------------------------------------------------------
__global__ void cvtT_kernel(const float* __restrict__ src, bf16* __restrict__ dst,
                            int K, int N) {
  const int i = blockIdx.x * blockDim.x + threadIdx.x;
  if (i < K * N) {
    const int k = i / N, n = i % N;
    dst[(size_t)n * K + k] = (bf16)src[(size_t)k * N + n];
  }
}

// ---------------------------------------------------------------------------
// Pass 1: gated projection. 16 tokens per WG, 4 waves = 4 x 16 output cols.
// vtrans=0 -> out[bh][s][dk]; vtrans=1 -> out[bh][dk][s] (pre-transposed V)
// ---------------------------------------------------------------------------
__global__ __launch_bounds__(128)
void qkv_kernel(const float* __restrict__ state, const float* __restrict__ Hm,
                const bf16* __restrict__ WeT, const float* __restrict__ be,
                const bf16* __restrict__ WhT, const float* __restrict__ bhp,
                const float* __restrict__ gE, const float* __restrict__ betaE,
                const float* __restrict__ gH, const float* __restrict__ betaH,
                bf16* __restrict__ out, int vtrans) {
  __shared__ __align__(16) bf16  eA[16][Dm];
  __shared__ __align__(16) bf16  hA[16][DFFsz];
  __shared__ __align__(16) float oE[16][Dm];
  __shared__ __align__(16) float oH[16][Dm];

  const int b = blockIdx.x, s0 = blockIdx.y * 16;
  const int tid = threadIdx.x, lane = tid & 31, wave = tid >> 5;

  // stage + convert inputs (float4 in, 4x bf16 out); pad rows -> 0
  for (int i = tid * 4; i < 16 * Dm; i += 128 * 4) {
    const int r = i / Dm, c = i % Dm, s = s0 + r;
    float4 f = (s < Ssz) ? *(const float4*)&state[((size_t)b * Ssz + s) * Dm + c]
                         : make_float4(0.f, 0.f, 0.f, 0.f);
    v4bf o; o[0] = (bf16)f.x; o[1] = (bf16)f.y; o[2] = (bf16)f.z; o[3] = (bf16)f.w;
    *(v4bf*)&eA[r][c] = o;
  }
  for (int i = tid * 4; i < 16 * DFFsz; i += 128 * 4) {
    const int r = i / DFFsz, c = i % DFFsz, s = s0 + r;
    float4 f = (s < Ssz) ? *(const float4*)&Hm[((size_t)b * Ssz + s) * DFFsz + c]
                         : make_float4(0.f, 0.f, 0.f, 0.f);
    v4bf o; o[0] = (bf16)f.x; o[1] = (bf16)f.y; o[2] = (bf16)f.z; o[3] = (bf16)f.w;
    *(v4bf*)&hA[r][c] = o;
  }
  __syncthreads();

  {
    const int nb = wave * 16;
    v8f accE = {0.f,0.f,0.f,0.f,0.f,0.f,0.f,0.f};
    v8f accH = {0.f,0.f,0.f,0.f,0.f,0.f,0.f,0.f};
#pragma unroll
    for (int ks = 0; ks < Dm; ks += 32)
      accE = wmma_bf16(load_a_frag(&eA[0][0], Dm, lane, ks),
                       load_bt_frag(WeT, Dm, lane, ks, nb), accE);
#pragma unroll
    for (int ks = 0; ks < DFFsz; ks += 32)
      accH = wmma_bf16(load_a_frag(&hA[0][0], DFFsz, lane, ks),
                       load_bt_frag(WhT, DFFsz, lane, ks, nb), accH);
    const int col = nb + (lane & 15);
#pragma unroll
    for (int v = 0; v < 8; ++v) {
      const int row = v + ((lane >> 4) << 3);
      oE[row][col] = accE[v] + be[col];
      oH[row][col] = accH[v] + bhp[col];
    }
  }
  __syncthreads();

  // row LayerNorm (D=64) with 8 lanes per row, gate, emit bf16
  {
    const int row = tid >> 3, sub = tid & 7;
    const int s = s0 + row;
    float sE = 0.f, sH = 0.f;
    for (int c = sub; c < Dm; c += 8) { sE += oE[row][c]; sH += oH[row][c]; }
#pragma unroll
    for (int o = 4; o; o >>= 1) { sE += __shfl_xor(sE, o, 8); sH += __shfl_xor(sH, o, 8); }
    const float mE = sE * (1.f / Dm), mH = sH * (1.f / Dm);
    float vE = 0.f, vH = 0.f;
    for (int c = sub; c < Dm; c += 8) {
      const float d0 = oE[row][c] - mE; vE += d0 * d0;
      const float d1 = oH[row][c] - mH; vH += d1 * d1;
    }
#pragma unroll
    for (int o = 4; o; o >>= 1) { vE += __shfl_xor(vE, o, 8); vH += __shfl_xor(vH, o, 8); }
    const float rE = rsqrtf(vE * (1.f / Dm) + 1e-5f);
    const float rH = rsqrtf(vH * (1.f / Dm) + 1e-5f);
    for (int c = sub; c < Dm; c += 8) {
      const float e = (oE[row][c] - mE) * rE * gE[c] + betaE[c];
      const float h = (oH[row][c] - mH) * rH * gH[c] + betaH[c];
      const float g = (s < Ssz) ? e * h : 0.f;
      const int head = c >> 5, dk = c & 31;
      const size_t base = ((size_t)b * NHn + head);
      if (vtrans)
        out[(base * DKn + dk) * SP + s] = (bf16)g;       // V^T: [dk][s]
      else
        out[(base * SP + s) * DKn + dk] = (bf16)g;       // [s][dk]
    }
  }
}

// ---------------------------------------------------------------------------
// Pass 2: attention for one (b,h) x 16-query tile; full-row f32 softmax.
// ---------------------------------------------------------------------------
__global__ __launch_bounds__(128)
void attn_kernel(const bf16* __restrict__ Q, const bf16* __restrict__ K,
                 const bf16* __restrict__ Vt, float* __restrict__ attn_out) {
  __shared__ __align__(16) bf16  qT[16][DKn];   //  1 KB
  __shared__ __align__(16) bf16  vTt[DKn][SP];  // 20 KB  (V^T: [dk][key])
  __shared__ __align__(16) float sc[16][SP];    // 20 KB
  __shared__ __align__(16) bf16  pr[16][SP];    // 10 KB

  const int bh = blockIdx.x, q0 = blockIdx.y * 16;
  const int tid = threadIdx.x, lane = tid & 31, wave = tid >> 5;

  const bf16* Qp = Q  + (size_t)bh * SP * DKn + (size_t)q0 * DKn;
  const bf16* Kp = K  + (size_t)bh * SP * DKn;
  const bf16* Vp = Vt + (size_t)bh * DKn * SP;
  __builtin_prefetch(Kp, 0, 1);  // global_prefetch_b8 on the K stream

  // async-stage q tile (1 KB) and V^T (20 KB) into LDS (ASYNCcnt)
  for (int c = tid; c < (16 * DKn * 2) / 16; c += 128)
    async_b128(lds_off(&qT[0][0]) + c * 16, (const char*)Qp + c * 16);
  for (int c = tid; c < (DKn * SP * 2) / 16; c += 128)
    async_b128(lds_off(&vTt[0][0]) + c * 16, (const char*)Vp + c * 16);
  wait_async0();
  __syncthreads();

  // scores = (q @ k^T)/sqrt(DK); K row-major [key][dk] IS B^T -> contiguous
  {
    const v16bf aq = load_a_frag(&qT[0][0], DKn, lane, 0);
    const float scale = 0.17677669529663687f;  // 1/sqrt(32)
    for (int kt = wave; kt < SP / 16; kt += 4) {
      v8f acc = {0.f,0.f,0.f,0.f,0.f,0.f,0.f,0.f};
      acc = wmma_bf16(aq, load_bt_frag(Kp, DKn, lane, 0, kt * 16), acc);
      const int col = kt * 16 + (lane & 15);
#pragma unroll
      for (int v = 0; v < 8; ++v)
        sc[v + ((lane >> 4) << 3)][col] = acc[v] * scale;
    }
  }
  __syncthreads();

  // softmax: 8 lanes per row, shfl_xor reductions
  {
    const int row = tid >> 3, sub = tid & 7;
    float m = -1e30f;
    for (int c = sub; c < Ssz; c += 8) m = fmaxf(m, sc[row][c]);
#pragma unroll
    for (int o = 4; o; o >>= 1) m = fmaxf(m, __shfl_xor(m, o, 8));
    float sum = 0.f;
    for (int c = sub; c < Ssz; c += 8) {
      const float e = __expf(sc[row][c] - m);
      sum += e; sc[row][c] = e;
    }
#pragma unroll
    for (int o = 4; o; o >>= 1) sum += __shfl_xor(sum, o, 8);
    const float inv = 1.f / sum;
    for (int c = sub; c < SP; c += 8)
      pr[row][c] = (c < Ssz) ? (bf16)(sc[row][c] * inv) : (bf16)0.f;
  }
  __syncthreads();

  // out = P @ V : N=DK=32 -> waves 0,1 ; K=SP=320 -> 10 WMMA steps
  if (wave < 2) {
    v8f acc = {0.f,0.f,0.f,0.f,0.f,0.f,0.f,0.f};
#pragma unroll
    for (int ks = 0; ks < SP; ks += 32)
      acc = wmma_bf16(load_a_frag(&pr[0][0], SP, lane, ks),
                      load_bt_frag(&vTt[0][0], SP, lane, ks, wave * 16), acc);
    const int batch = bh / NHn, head = bh % NHn;
    const int col = wave * 16 + (lane & 15);
#pragma unroll
    for (int v = 0; v < 8; ++v) {
      const int row = v + ((lane >> 4) << 3);
      const int s = q0 + row;
      if (s < Ssz)
        attn_out[((size_t)batch * Ssz + s) * Dm + head * DKn + col] = acc[v];
    }
  }
}

// ---------------------------------------------------------------------------
// Pass 3: Z1 = state+attn; LN; GELU(Z2@W1+b1)@W2+b2; out = Z1+Z4
// W1^T panel (256x64 bf16, 32 KB contiguous) staged by the Tensor Data Mover.
// ---------------------------------------------------------------------------
__global__ __launch_bounds__(256)
void ffn_kernel(const float* __restrict__ state, const float* __restrict__ attn_out,
                const float* __restrict__ g1, const float* __restrict__ beta1,
                const bf16* __restrict__ W1T, const float* __restrict__ b1,
                const bf16* __restrict__ W2T, const float* __restrict__ b2,
                float* __restrict__ out) {
  __shared__ __align__(16) float z1[16][Dm];        //  4 KB (final residual)
  __shared__ __align__(16) bf16  z2[16][Dm];        //  2 KB
  __shared__ __align__(16) bf16  w1T[DFFsz][Dm];    // 32 KB  (W1^T [N][K])
  __shared__ __align__(16) bf16  z3[16][DFFsz];     //  8 KB

  const int b = blockIdx.x, s0 = blockIdx.y * 16;
  const int tid = threadIdx.x, lane = tid & 31, wave = tid >> 5;

  // TDM: one descriptor, wave 0 issues; overlaps with Z1 staging below.
  if (wave == 0)
    tdm_load_2d_bf16(lds_off(&w1T[0][0]), W1T,
                     /*dim0=*/Dm, /*dim1=*/DFFsz, /*stride0=*/Dm);

  for (int i = tid * 4; i < 16 * Dm; i += 256 * 4) {
    const int r = i / Dm, c = i % Dm, s = s0 + r;
    if (s < Ssz) {
      const float4 a = *(const float4*)&state[((size_t)b * Ssz + s) * Dm + c];
      const float4 o = *(const float4*)&attn_out[((size_t)b * Ssz + s) * Dm + c];
      *(float4*)&z1[r][c] = make_float4(a.x + o.x, a.y + o.y, a.z + o.z, a.w + o.w);
    } else {
      *(float4*)&z1[r][c] = make_float4(0.f, 0.f, 0.f, 0.f);
    }
  }
  if (wave == 0) __builtin_amdgcn_s_wait_tensorcnt(0);
  __syncthreads();

  // LayerNorm: 16 lanes per row
  {
    const int row = tid >> 4, sub = tid & 15;
    float sm = 0.f;
    for (int c = sub; c < Dm; c += 16) sm += z1[row][c];
#pragma unroll
    for (int o = 8; o; o >>= 1) sm += __shfl_xor(sm, o, 16);
    const float m = sm * (1.f / Dm);
    float vv = 0.f;
    for (int c = sub; c < Dm; c += 16) { const float d = z1[row][c] - m; vv += d * d; }
#pragma unroll
    for (int o = 8; o; o >>= 1) vv += __shfl_xor(vv, o, 16);
    const float r = rsqrtf(vv * (1.f / Dm) + 1e-5f);
    for (int c = sub; c < Dm; c += 16)
      z2[row][c] = (bf16)((z1[row][c] - m) * r * g1[c] + beta1[c]);
  }
  __syncthreads();

  // GEMM1 + exact-erf GELU: 16 n-tiles over 8 waves, K=64 (2 steps)
  for (int nt = wave; nt < DFFsz / 16; nt += 8) {
    v8f acc = {0.f,0.f,0.f,0.f,0.f,0.f,0.f,0.f};
#pragma unroll
    for (int ks = 0; ks < Dm; ks += 32)
      acc = wmma_bf16(load_a_frag(&z2[0][0], Dm, lane, ks),
                      load_bt_frag(&w1T[0][0], Dm, lane, ks, nt * 16), acc);
    const int col = nt * 16 + (lane & 15);
    const float bias = b1[col];
#pragma unroll
    for (int v = 0; v < 8; ++v) {
      const int row = v + ((lane >> 4) << 3);
      const float x = acc[v] + bias;
      z3[row][col] = (bf16)(0.5f * x * (1.f + erff(x * 0.70710678118654752f)));
    }
  }
  __syncthreads();

  // GEMM2: 4 n-tiles (waves 0-3), K=256 (8 steps); W2^T streamed from L2
  if (wave < 4) {
    v8f acc = {0.f,0.f,0.f,0.f,0.f,0.f,0.f,0.f};
#pragma unroll
    for (int ks = 0; ks < DFFsz; ks += 32)
      acc = wmma_bf16(load_a_frag(&z3[0][0], DFFsz, lane, ks),
                      load_bt_frag(W2T, DFFsz, lane, ks, wave * 16), acc);
    const int col = wave * 16 + (lane & 15);
    const float bias = b2[col];
#pragma unroll
    for (int v = 0; v < 8; ++v) {
      const int row = v + ((lane >> 4) << 3);
      const int s = s0 + row;
      if (s < Ssz)
        out[((size_t)b * Ssz + s) * Dm + col] = z1[row][col] + acc[v] + bias;
    }
  }
}

// ---------------------------------------------------------------------------
extern "C" void kernel_launch(void* const* d_in, const int* in_sizes, int n_in,
                              void* d_out, int out_size, void* d_ws, size_t ws_size,
                              hipStream_t stream) {
  const float* state  = (const float*)d_in[0];
  const float* Hm     = (const float*)d_in[1];
  const float* Wq     = (const float*)d_in[2];
  const float* bq     = (const float*)d_in[3];
  const float* Wk     = (const float*)d_in[4];
  const float* bk     = (const float*)d_in[5];
  const float* Wv     = (const float*)d_in[6];
  const float* bv     = (const float*)d_in[7];
  const float* Wcq    = (const float*)d_in[8];
  const float* bcq    = (const float*)d_in[9];
  const float* Wck    = (const float*)d_in[10];
  const float* bck    = (const float*)d_in[11];
  const float* Wcv    = (const float*)d_in[12];
  const float* bcv    = (const float*)d_in[13];
  const float* gQE    = (const float*)d_in[14];
  const float* betaQE = (const float*)d_in[15];
  const float* gKE    = (const float*)d_in[16];
  const float* betaKE = (const float*)d_in[17];
  const float* gVE    = (const float*)d_in[18];
  const float* betaVE = (const float*)d_in[19];
  const float* gQH    = (const float*)d_in[20];
  const float* betaQH = (const float*)d_in[21];
  const float* gKH    = (const float*)d_in[22];
  const float* betaKH = (const float*)d_in[23];
  const float* gVH    = (const float*)d_in[24];
  const float* betaVH = (const float*)d_in[25];
  const float* W1     = (const float*)d_in[26];
  const float* b1     = (const float*)d_in[27];
  const float* W2     = (const float*)d_in[28];
  const float* b2     = (const float*)d_in[29];
  const float* g1     = (const float*)d_in[30];
  const float* beta1  = (const float*)d_in[31];

  // ---- workspace layout (256B aligned) ----
  size_t off = 0;
  auto take = [&](size_t bytes) {
    size_t cur = off;
    off = (off + bytes + 255) & ~(size_t)255;
    return cur;
  };
  char* ws = (char*)d_ws;
  const size_t WSM = 64 * 64, WCB = 256 * 64, W1E = 64 * 256, W2E = 256 * 64;
  bf16* WqT  = (bf16*)(ws + take(WSM * 2));   // [64][64]   (N,K)
  bf16* WkT  = (bf16*)(ws + take(WSM * 2));
  bf16* WvT  = (bf16*)(ws + take(WSM * 2));
  bf16* WcqT = (bf16*)(ws + take(WCB * 2));   // [64][256]
  bf16* WckT = (bf16*)(ws + take(WCB * 2));
  bf16* WcvT = (bf16*)(ws + take(WCB * 2));
  bf16* W1Tp = (bf16*)(ws + take(W1E * 2));   // [256][64]
  bf16* W2Tp = (bf16*)(ws + take(W2E * 2));   // [64][256]
  const size_t QKVE = (size_t)Bsz * NHn * SP * DKn;
  bf16*  Qb = (bf16*)(ws + take(QKVE * 2));   // [bh][SP][DK]
  bf16*  Kb = (bf16*)(ws + take(QKVE * 2));   // [bh][SP][DK]
  bf16*  Vt = (bf16*)(ws + take(QKVE * 2));   // [bh][DK][SP]  (pre-transposed)
  float* aO = (float*)(ws + take((size_t)Bsz * Ssz * Dm * 4));
  (void)ws_size; (void)n_in; (void)in_sizes; (void)out_size;

  // ---- pass 0: weights -> bf16, transposed to [N][K] ----
  auto cvt = [&](const float* s, bf16* d, int K, int N) {
    const size_t n = (size_t)K * N;
    cvtT_kernel<<<dim3((unsigned)((n + 255) / 256)), 256, 0, stream>>>(s, d, K, N);
  };
  cvt(Wq, WqT, 64, 64);    cvt(Wk, WkT, 64, 64);    cvt(Wv, WvT, 64, 64);
  cvt(Wcq, WcqT, 256, 64); cvt(Wck, WckT, 256, 64); cvt(Wcv, WcvT, 256, 64);
  cvt(W1, W1Tp, 64, 256);  cvt(W2, W2Tp, 256, 64);

  // ---- pass 1: gated q/k/v projections ----
  const dim3 gProj(Bsz, SP / 16);
  qkv_kernel<<<gProj, 128, 0, stream>>>(state, Hm, WqT, bq, WcqT, bcq,
                                        gQE, betaQE, gQH, betaQH, Qb, 0);
  qkv_kernel<<<gProj, 128, 0, stream>>>(state, Hm, WkT, bk, WckT, bck,
                                        gKE, betaKE, gKH, betaKH, Kb, 0);
  qkv_kernel<<<gProj, 128, 0, stream>>>(state, Hm, WvT, bv, WcvT, bcv,
                                        gVE, betaVE, gVH, betaVH, Vt, 1);

  // ---- pass 2: attention ----
  attn_kernel<<<dim3(Bsz * NHn, SP / 16), 128, 0, stream>>>(Qb, Kb, Vt, aO);

  // ---- pass 3: residual + LN + FFN + residual ----
  ffn_kernel<<<dim3(Bsz, (Ssz + 15) / 16), 256, 0, stream>>>(
      state, aO, g1, beta1, W1Tp, b1, W2Tp, b2, (float*)d_out);
}
